// EnrichClassifier_77927886619274
// MI455X (gfx1250) — matrix operand order repaired and checked by reading, not compiled, not run on hardware
//
#include <hip/hip_runtime.h>

typedef _Float16 half_t;
typedef __attribute__((ext_vector_type(16))) _Float16 v16h;
typedef __attribute__((ext_vector_type(8)))  float    v8f;

struct U4 { unsigned x, y, z, w; };

union Frag { v16h v; U4 q[2]; };

// ---------------------------------------------------------------------------
// Fold eval-mode BatchNorm + bias into per-column scale/shift: y = z*s + t
// ---------------------------------------------------------------------------
__global__ __launch_bounds__(256)
void bn_fold_kernel(const float* __restrict__ gamma, const float* __restrict__ beta,
                    const float* __restrict__ rm, const float* __restrict__ rv,
                    const float* __restrict__ bias,
                    float* __restrict__ s, float* __restrict__ t, int n)
{
    int i = blockIdx.x * 256 + threadIdx.x;
    if (i >= n) return;
    float sc = gamma[i] * rsqrtf(rv[i] + 1e-5f);
    s[i] = sc;
    t[i] = (bias[i] - rm[i]) * sc + beta[i];
}

__global__ __launch_bounds__(256)
void bias_fold_kernel(const float* __restrict__ bias,
                      float* __restrict__ s, float* __restrict__ t, int n)
{
    int i = blockIdx.x * 256 + threadIdx.x;
    if (i >= n) return;
    s[i] = 1.0f;
    t[i] = bias[i];
}

// ---------------------------------------------------------------------------
// Pack masked weight w∘m ([N,K] row-major fp32) into f16 WMMA B-fragment
// order: frag(kt,nt) = 512 contiguous f16, lane-major; element i of lane
// (hi*16+nl) holds (K = kt*32 + (i<8?0:16) + hi*8 + (i&7), N = nt*16 + nl). Zero-padded.
// ---------------------------------------------------------------------------
__global__ __launch_bounds__(256)
void pack_w_kernel(const float* __restrict__ w, const float* __restrict__ m,
                   int K, int N, int NT, half_t* __restrict__ out, long total)
{
    long idx = (long)blockIdx.x * 256 + threadIdx.x;
    if (idx >= total) return;
    int  e    = (int)(idx & 511);
    long frag = idx >> 9;
    int  nt   = (int)(frag % NT);
    int  kt   = (int)(frag / NT);
    int  lane = e >> 4;
    int  i    = e & 15;
    int  k = kt * 32 + ((i & 8) << 1) + (lane >> 4) * 8 + (i & 7);
    int  n = nt * 16 + (lane & 15);
    float v = 0.0f;
    if (k < K && n < N) {
        size_t o = (size_t)n * K + k;
        v = m ? (w[o] * m[o]) : w[o];
    }
    out[idx] = (half_t)v;
}

// ---------------------------------------------------------------------------
// Pack fp32 x ([rows,K] row-major) into f16 WMMA A-fragment order:
// frag(rowTile,kt) = 512 contiguous f16, lane-major; lane (hi*16+m), element
// e (pair p=e>>1) holds (row = rowTile*16+m,
//                        K = kt*32 + (p<4?0:16) + hi*8 + (p&3)*2 + (e&1)).
// ---------------------------------------------------------------------------
__global__ __launch_bounds__(256)
void pack_a_kernel(const float* __restrict__ x, int K, int KT,
                   half_t* __restrict__ out, long total)
{
    long idx = (long)blockIdx.x * 256 + threadIdx.x;
    if (idx >= total) return;
    int  e    = (int)(idx & 15);
    long t2   = idx >> 4;
    int  lane = (int)(t2 & 31);
    long frag = t2 >> 5;
    int  kt   = (int)(frag % KT);
    long rowTile = frag / KT;
    int  m  = lane & 15;
    int  hi = lane >> 4;
    int  p  = e >> 1;
    int  k  = kt * 32 + ((p & 4) << 2) + hi * 8 + (p & 3) * 2 + (e & 1);
    float v = 0.0f;
    if (k < K) v = x[(rowTile * 16 + m) * (long)K + k];
    out[idx] = (half_t)v;
}

// ---------------------------------------------------------------------------
// WMMA GEMM, fully fragment-packed I/O. No LDS, no barriers.
//   Block: 256 threads = 8 waves as 4(row) x 2(col) -> 128x128 tile.
//   Wave tile: 32x64 = 2(mi) x 4(ni) fragments, 8 WMMA per k-step.
//   Ap: A fragments [rowTile][kt][512], Wp: B fragments [kt][nt][512].
//   PACKOUT: store f16 result in A-fragment order for the next layer
//            (zeroing columns in [N, Nstore) = next layer's K padding);
//   else: row-major fp32 store (final layer).
// ---------------------------------------------------------------------------
template <bool PACKOUT, bool RELU>
__global__ __launch_bounds__(256)
void gemm_wmma(const half_t* __restrict__ Ap, int KT,
               const half_t* __restrict__ Wp, int NT,
               const float* __restrict__ scale, const float* __restrict__ shift,
               half_t* __restrict__ OutP, int KTn,
               float* __restrict__ OutF, int ldo,
               int N, int Nstore)
{
    const int tid     = threadIdx.x;
    const int lane    = tid & 31;
    const int wid     = tid >> 5;       // 0..7
    const int waveRow = wid >> 1;       // 0..3
    const int waveCol = wid & 1;        // 0..1
    const int colBlk  = blockIdx.x * 128;
    const int rtBase  = blockIdx.y * 8 + waveRow * 2;   // rowTile of mi=0
    const int hiC     = lane >> 4;
    const int lm      = lane & 15;

    v8f acc[2][4] = {};

    const half_t* aPtr0 = Ap + ((size_t)rtBase * KT) * 512 + (size_t)lane * 16;
    const half_t* aPtr1 = aPtr0 + (size_t)KT * 512;
    const half_t* bPtr  = Wp + ((size_t)(colBlk >> 4) + waveCol * 4) * 512 + (size_t)lane * 16;
    const size_t  bStep = (size_t)NT * 512;

    for (int kt = 0; kt < KT; ++kt) {
        Frag a0, a1;
        {
            const U4* p0 = (const U4*)aPtr0;
            const U4* p1 = (const U4*)aPtr1;
            a0.q[0] = p0[0]; a0.q[1] = p0[1];
            a1.q[0] = p1[0]; a1.q[1] = p1[1];
        }
        Frag b[4];
        #pragma unroll
        for (int ni = 0; ni < 4; ++ni) {
            const U4* p = (const U4*)(bPtr + (size_t)ni * 512);
            b[ni].q[0] = p[0];
            b[ni].q[1] = p[1];
        }
        #pragma unroll
        for (int ni = 0; ni < 4; ++ni) {
            acc[0][ni] = __builtin_amdgcn_wmma_f32_16x16x32_f16(
                false, a0.v, false, b[ni].v, (short)0, acc[0][ni], false, false);
            acc[1][ni] = __builtin_amdgcn_wmma_f32_16x16x32_f16(
                false, a1.v, false, b[ni].v, (short)0, acc[1][ni], false, false);
        }
        aPtr0 += 512;
        aPtr1 += 512;
        bPtr  += bStep;
    }

    // ---- epilogue: scale/shift (+ReLU), write packed-f16 or row-major f32 ----
    #pragma unroll
    for (int ni = 0; ni < 4; ++ni) {
        const int col = colBlk + waveCol * 64 + ni * 16 + lm;
        if (col >= Nstore) continue;
        const bool live = col < N;
        const float s = live ? scale[col] : 0.0f;
        const float t = live ? shift[col] : 0.0f;
        if constexpr (PACKOUT) {
            // col -> (kt, lane-half, element) of the next layer's A fragment
            const int ktO = col >> 5;
            const int kc  = col & 31;
            const int hiA = (kc >> 3) & 1;
            const int pA  = ((kc & 16) >> 2) | ((kc & 7) >> 1);
            const int eA  = 2 * pA + (kc & 1);
            #pragma unroll
            for (int mi = 0; mi < 2; ++mi) {
                const size_t fragBase = ((size_t)(rtBase + mi) * KTn + ktO) * 512;
                half_t* o = OutP + fragBase + (size_t)(hiA * 16 + hiC * 8) * 16 + eA;
                #pragma unroll
                for (int r = 0; r < 8; ++r) {
                    float v = acc[mi][ni][r] * s + t;
                    if (RELU && v < 0.0f) v = 0.0f;
                    if (!live) v = 0.0f;
                    o[(size_t)r * 16] = (half_t)v;
                }
            }
        } else {
            if (!live) continue;
            #pragma unroll
            for (int mi = 0; mi < 2; ++mi) {
                const int rowBase = (rtBase + mi) * 16 + hiC * 8;
                #pragma unroll
                for (int r = 0; r < 8; ++r) {
                    float v = acc[mi][ni][r] * s + t;
                    if (RELU && v < 0.0f) v = 0.0f;
                    OutF[(size_t)(rowBase + r) * ldo + col] = v;
                }
            }
        }
    }
}

// ---------------------------------------------------------------------------
extern "C" void kernel_launch(void* const* d_in, const int* in_sizes, int n_in,
                              void* d_out, int out_size, void* d_ws, size_t ws_size,
                              hipStream_t stream)
{
    (void)in_sizes; (void)n_in; (void)out_size;

    const float* x   = (const float*)d_in[0];
    const float* w1  = (const float*)d_in[1];
    const float* b1  = (const float*)d_in[2];
    const float* m1  = (const float*)d_in[3];
    const float* w2  = (const float*)d_in[4];
    const float* b2  = (const float*)d_in[5];
    const float* m2  = (const float*)d_in[6];
    const float* w3  = (const float*)d_in[7];
    const float* b3  = (const float*)d_in[8];
    const float* m3  = (const float*)d_in[9];
    const float* w4  = (const float*)d_in[10];
    const float* b4  = (const float*)d_in[11];
    const float* m4  = (const float*)d_in[12];
    const float* g1  = (const float*)d_in[13];
    const float* be1 = (const float*)d_in[14];
    const float* rm1 = (const float*)d_in[15];
    const float* rv1 = (const float*)d_in[16];
    const float* g2  = (const float*)d_in[17];
    const float* be2 = (const float*)d_in[18];
    const float* rm2 = (const float*)d_in[19];
    const float* rv2 = (const float*)d_in[20];
    const float* g3  = (const float*)d_in[21];
    const float* be3 = (const float*)d_in[22];
    const float* rm3 = (const float*)d_in[23];
    const float* rv3 = (const float*)d_in[24];
    const float* wc  = (const float*)d_in[25];
    const float* bc  = (const float*)d_in[26];
    float* out = (float*)d_out;

    // dims (real)
    const int K1 = 5000, N1 = 4000;
    const int K2 = 4000, N2 = 2000;
    const int K3 = 2000, N3 = 1000;
    const int K4 = 1000, N4 = 200;
    const int K5 = 200,  N5 = 50;
    // K-tile counts (padded K / 32) and B fragment-column counts (padded N / 16)
    const int KT1 = 157, NT1 = 256;     // K pad 5024, N pad 4096
    const int KT2 = 125, NT2 = 128;     // 4000 exact, N pad 2048
    const int KT3 = 63,  NT3 = 64;      // K pad 2016, N pad 1024
    const int KT4 = 32,  NT4 = 16;      // K pad 1024, N pad 256
    const int KT5 = 7,   NT5 = 8;       // K pad 224,  N pad 128
    const int RT  = 512;                // 8192 rows / 16

    auto al = [](size_t b) -> size_t { return (b + 255) & ~(size_t)255; };
    char* ws = (char*)d_ws;

    // small scale/shift region
    size_t off = 0;
    auto take = [&](size_t bytes) -> size_t { size_t o = off; off += al(bytes); return o; };
    float* s1 = (float*)(ws + take(4 * N1));
    float* t1 = (float*)(ws + take(4 * N1));
    float* s2 = (float*)(ws + take(4 * N2));
    float* t2 = (float*)(ws + take(4 * N2));
    float* s3 = (float*)(ws + take(4 * N3));
    float* t3 = (float*)(ws + take(4 * N3));
    float* s4 = (float*)(ws + take(4 * N4));
    float* t4 = (float*)(ws + take(4 * N4));
    float* s5 = (float*)(ws + take(4 * 64));
    float* t5 = (float*)(ws + take(4 * 64));

    // big regions with lifetime-based reuse (packs interleave with GEMMs)
    const long TA1 = (long)RT * KT1 * 512;           // packed x
    const long TW1 = (long)KT1 * NT1 * 512;
    const long TW2 = (long)KT2 * NT2 * 512;
    const long TW3 = (long)KT3 * NT3 * 512;
    const long TW4 = (long)KT4 * NT4 * 512;
    const long TW5 = (long)KT5 * NT5 * 512;
    const long TH1 = (long)RT * KT2 * 512;           // h1 packed (width 4000)
    const long TH2 = (long)RT * KT3 * 512;           // h2 packed (width 2016)
    const long TH3 = (long)RT * KT4 * 512;           // h3 packed (width 1024)
    const long TH4 = (long)RT * KT5 * 512;           // h4 packed (width 224)

    const size_t oA = take(2 * (size_t)TA1);         // 82.3 MB: xp | {W2p, h2, W5p}
    const size_t oB = take(2 * (size_t)TW1);         // 41.2 MB: W1p | {W3p, h3}
    const size_t oC = take(2 * (size_t)TH1);         // 65.5 MB: h1  | {W4p, h4}
    if (off > ws_size) return;                       // workspace too small

    half_t* xp  = (half_t*)(ws + oA);
    half_t* W2p = (half_t*)(ws + oA);                            // after L1
    half_t* h2  = (half_t*)(ws + oA + al(2 * (size_t)TW2));
    half_t* W5p = (half_t*)(ws + oA + al(2 * (size_t)TW2) + al(2 * (size_t)TH2));
    half_t* W1p = (half_t*)(ws + oB);
    half_t* W3p = (half_t*)(ws + oB);                            // after L1
    half_t* h3  = (half_t*)(ws + oB + al(2 * (size_t)TW3));
    half_t* h1  = (half_t*)(ws + oC);
    half_t* W4p = (half_t*)(ws + oC);                            // after L2
    half_t* h4  = (half_t*)(ws + oC + al(2 * (size_t)TW4));

    // 1) fold BN / bias into scale+shift
    bn_fold_kernel<<<(N1 + 255) / 256, 256, 0, stream>>>(g1, be1, rm1, rv1, b1, s1, t1, N1);
    bn_fold_kernel<<<(N2 + 255) / 256, 256, 0, stream>>>(g2, be2, rm2, rv2, b2, s2, t2, N2);
    bn_fold_kernel<<<(N3 + 255) / 256, 256, 0, stream>>>(g3, be3, rm3, rv3, b3, s3, t3, N3);
    bias_fold_kernel<<<1, 256, 0, stream>>>(b4, s4, t4, N4);
    bias_fold_kernel<<<1, 256, 0, stream>>>(bc, s5, t5, N5);

    // 2) pack x and layer-1 weights, then alternate GEMM / pack for reuse
    pack_a_kernel<<<(unsigned)((TA1 + 255) / 256), 256, 0, stream>>>(x, K1, KT1, xp, TA1);
    pack_w_kernel<<<(unsigned)((TW1 + 255) / 256), 256, 0, stream>>>(w1, m1, K1, N1, NT1, W1p, TW1);
    gemm_wmma<true, true><<<dim3(32, 64), 256, 0, stream>>>(
        xp, KT1, W1p, NT1, s1, t1, h1, KT2, nullptr, 0, N1, KT2 * 32);

    pack_w_kernel<<<(unsigned)((TW2 + 255) / 256), 256, 0, stream>>>(w2, m2, K2, N2, NT2, W2p, TW2);
    gemm_wmma<true, true><<<dim3(16, 64), 256, 0, stream>>>(
        h1, KT2, W2p, NT2, s2, t2, h2, KT3, nullptr, 0, N2, KT3 * 32);

    pack_w_kernel<<<(unsigned)((TW3 + 255) / 256), 256, 0, stream>>>(w3, m3, K3, N3, NT3, W3p, TW3);
    gemm_wmma<true, true><<<dim3(8, 64), 256, 0, stream>>>(
        h2, KT3, W3p, NT3, s3, t3, h3, KT4, nullptr, 0, N3, KT4 * 32);

    pack_w_kernel<<<(unsigned)((TW4 + 255) / 256), 256, 0, stream>>>(w4, m4, K4, N4, NT4, W4p, TW4);
    gemm_wmma<true, true><<<dim3(2, 64), 256, 0, stream>>>(
        h3, KT4, W4p, NT4, s4, t4, h4, KT5, nullptr, 0, N4, KT5 * 32);

    pack_w_kernel<<<(unsigned)((TW5 + 255) / 256), 256, 0, stream>>>(wc, nullptr, K5, N5, NT5, W5p, TW5);
    gemm_wmma<false, false><<<dim3(1, 64), 256, 0, stream>>>(
        h4, KT5, W5p, NT5, s5, t5, nullptr, 0, out, N5, N5, N5);
}
